// MultiHeadAttentionChannel_2508260900976
// MI455X (gfx1250) — matrix operand order repaired and checked by reading, not compiled, not run on hardware
//
#include <hip/hip_runtime.h>
#include <hip/hip_bf16.h>
#include <math.h>

// ---------------------------------------------------------------------------
// CDNA5 (gfx1250) channel-attention: bf16 WMMA (V_WMMA_F32_16X16X32_BF16),
// double-buffered GLOBAL_LOAD_ASYNC_TO_LDS_B128 staging (ASYNCcnt pipeline),
// split-K scores with global_atomic_add_f32.
// Shapes fixed by the reference: B=4, S=4096, HID=1024, H=8, D=128.
// ---------------------------------------------------------------------------

typedef __attribute__((ext_vector_type(16))) __bf16 v16bf;
typedef __attribute__((ext_vector_type(8)))  float v8f;

#define BATCH   4
#define SEQ     4096
#define HID     1024
#define NHEAD   8
#define DCH     128
#define MROWS   (BATCH * SEQ)          // 16384
#define LPITCH  40                     // LDS row pitch in bf16 elems (80B: 16B-aligned rows)

__device__ __forceinline__ unsigned short f2bf(float f) {
    unsigned u = __float_as_uint(f);
    unsigned r = u + 0x7FFFu + ((u >> 16) & 1u);   // round-to-nearest-even
    return (unsigned short)(r >> 16);
}

// Packed f32x2 -> bf16x2 (one VALU op when the cvt_pk builtin exists).
__device__ __forceinline__ unsigned pkbf(float a, float b) {
#if __has_builtin(__builtin_amdgcn_cvt_pk_bf16_f32)
    typedef __attribute__((ext_vector_type(2))) __bf16 v2bf;
    union { v2bf v; unsigned u; } x;
    x.v = __builtin_amdgcn_cvt_pk_bf16_f32(a, b);
    return x.u;
#else
    return (unsigned)f2bf(a) | ((unsigned)f2bf(b) << 16);
#endif
}

// Async 16B global->LDS copy (tracked by ASYNCcnt).
__device__ __forceinline__ void async_cp16(unsigned lds_off, const void* g) {
    asm volatile("global_load_async_to_lds_b128 %0, %1, off"
                 :: "v"(lds_off), "v"(g) : "memory");
}
__device__ __forceinline__ void wait_async_all() {
#if __has_builtin(__builtin_amdgcn_s_wait_asynccnt)
    __builtin_amdgcn_s_wait_asynccnt(0);
#else
    asm volatile("s_wait_asynccnt 0" ::: "memory");
#endif
}
// Allow the newest 4 per-wave async ops to remain in flight (the next stage).
__device__ __forceinline__ void wait_async_prev() {
#if __has_builtin(__builtin_amdgcn_s_wait_asynccnt)
    __builtin_amdgcn_s_wait_asynccnt(4);
#else
    asm volatile("s_wait_asynccnt 4" ::: "memory");
#endif
}

union FragBF { uint4 q[2]; v16bf v; };

// A fragment: 16x32 bf16 tile, row-major in LDS (rows = M, contiguous K).
// ISA: lanes 0-15 (M=lane) hold K=0..7,16..23; lanes 16-31 hold K=8..15,24..31.
__device__ __forceinline__ v16bf load_fragA(const unsigned short* lds) {
    int lane = threadIdx.x & 31;
    const unsigned short* row = lds + (lane & 15) * LPITCH;
    int koff = (lane >> 4) << 3;                   // 0 or 8
    FragBF f;
    f.q[0] = *(const uint4*)(row + koff);
    f.q[1] = *(const uint4*)(row + koff + 16);
    return f.v;
}

// B fragment: 32x16 (KxN) tile stored TRANSPOSED in LDS as Bt[n][k].
// ISA: lanes 0-15 (N=lane) hold K=0..7,16..23; lanes 16-31 hold K=8..15,24..31.
__device__ __forceinline__ v16bf load_fragB(const unsigned short* lds) {
    int lane = threadIdx.x & 31;
    const unsigned short* row = lds + (lane & 15) * LPITCH + ((lane >> 4) << 3);
    FragBF f;
    f.q[0] = *(const uint4*)(row);
    f.q[1] = *(const uint4*)(row + 16);
    return f.v;
}

#define WMMA_BF16(a, b, c) \
    __builtin_amdgcn_wmma_f32_16x16x32_bf16(false, (a), false, (b), (short)0, (c), false, false)

// Stage a 128x32 bf16 tile (row stride = ld elems) into LDS via async b128
// copies: 512 16B chunks, 2 per thread = 2 async instructions per wave.
__device__ __forceinline__ void stage_async(unsigned lds_base,
                                            const unsigned short* src,
                                            size_t ld, int tid) {
    #pragma unroll
    for (int e = 0; e < 2; ++e) {
        int idx = e * 256 + tid;                   // 0..511
        int r = idx >> 2;                          // 0..127
        int c = (idx & 3) * 8;                     // 0,8,16,24
        async_cp16(lds_base + (unsigned)(r * LPITCH + c) * 2,
                   src + (size_t)r * ld + c);
    }
}

// ---------------------------------------------------------------------------
// Kernel: streaming f32 -> bf16 conversion (float4 in, uint2 out)
// ---------------------------------------------------------------------------
__global__ __launch_bounds__(256) void cvt_bf16(const float* __restrict__ src,
                                                unsigned short* __restrict__ dst,
                                                long n4) {
    long i = (long)blockIdx.x * 256 + threadIdx.x;
    if (i >= n4) return;
    float4 f = ((const float4*)src)[i];
    uint2 p; p.x = pkbf(f.x, f.y); p.y = pkbf(f.z, f.w);
    ((uint2*)dst)[i] = p;
}

// ---------------------------------------------------------------------------
// Kernel: zero the attn accumulator region (float4 stores)
// ---------------------------------------------------------------------------
__global__ __launch_bounds__(256) void zero_f32(float* __restrict__ p, long n4) {
    long i = (long)blockIdx.x * 256 + threadIdx.x;
    if (i >= n4) return;
    ((float4*)p)[i] = make_float4(0.f, 0.f, 0.f, 0.f);
}

// ---------------------------------------------------------------------------
// Kernel: bf16 GEMM  Y[M,N] = A[M,K] @ Bm[N,K]^T
// Block tile 128x128, K-step 32, 8 waves (4Mx2N), wave 32x64 -> 8 WMMA/step.
// Both operands staged with double-buffered async global->LDS copies:
// issue stage(k+1) into buf^1, s_wait_asynccnt 4 (prev stage done, new one
// in flight behind the WMMAs), barrier, compute buf.
// ---------------------------------------------------------------------------
template <bool OUT_BF16>
__global__ __launch_bounds__(256) void gemm_bf16(const unsigned short* __restrict__ A,
                                                 const unsigned short* __restrict__ Bm,
                                                 void* __restrict__ Yv,
                                                 int M, int N, int K) {
    __shared__ __align__(16) unsigned short lA[2][128 * LPITCH];
    __shared__ __align__(16) unsigned short lB[2][128 * LPITCH];

    const int tid  = threadIdx.x;
    const int wave = tid >> 5;
    const int m0   = blockIdx.y * 128;
    const int n0   = blockIdx.x * 128;
    const int wm   = (wave & 3) * 32;
    const int wn   = (wave >> 2) * 64;
    const unsigned aBase[2] = {(unsigned)(size_t)(void*)lA[0], (unsigned)(size_t)(void*)lA[1]};
    const unsigned bBase[2] = {(unsigned)(size_t)(void*)lB[0], (unsigned)(size_t)(void*)lB[1]};
    const unsigned short* Arow = A + (size_t)m0 * K;
    const unsigned short* Brow = Bm + (size_t)n0 * K;

    v8f acc[2][4] = {};

    stage_async(aBase[0], Arow, (size_t)K, tid);     // prologue: buf 0
    stage_async(bBase[0], Brow, (size_t)K, tid);

    int buf = 0;
    for (int k0 = 0; k0 < K; k0 += 32) {
        if (k0 + 32 < K) {                           // prefetch next into buf^1
            stage_async(aBase[buf ^ 1], Arow + (k0 + 32), (size_t)K, tid);
            stage_async(bBase[buf ^ 1], Brow + (k0 + 32), (size_t)K, tid);
            wait_async_prev();                       // current buf's copies done
        } else {
            wait_async_all();
        }
        __syncthreads();

        v16bf aF[2], bF[4];
        #pragma unroll
        for (int i = 0; i < 2; ++i) aF[i] = load_fragA(&lA[buf][(wm + i * 16) * LPITCH]);
        #pragma unroll
        for (int j = 0; j < 4; ++j) bF[j] = load_fragB(&lB[buf][(wn + j * 16) * LPITCH]);
        #pragma unroll
        for (int i = 0; i < 2; ++i)
            #pragma unroll
            for (int j = 0; j < 4; ++j)
                acc[i][j] = WMMA_BF16(aF[i], bF[j], acc[i][j]);
        __syncthreads();                             // buf fully read before restage
        buf ^= 1;
    }

    const int lane = tid & 31;
    const int col = lane & 15;
    const int rowAdd = (lane >> 4) << 3;             // lanes 16-31 hold M = r+8
    #pragma unroll
    for (int i = 0; i < 2; ++i)
        #pragma unroll
        for (int j = 0; j < 4; ++j)
            #pragma unroll
            for (int r = 0; r < 8; ++r) {
                size_t m = m0 + wm + i * 16 + rowAdd + r;
                size_t n = n0 + wn + j * 16 + col;
                float v = acc[i][j][r];
                if (OUT_BF16) ((unsigned short*)Yv)[m * N + n] = f2bf(v);
                else          ((float*)Yv)[m * N + n] = v;
            }
}

// ---------------------------------------------------------------------------
// Kernel: split-K scores. scores[bh,i,j] += sum_{s in slice} qh[..i]*kh[..j]
// Grid (bh, 8 K-slices of 512); partials merged with global_atomic_add_f32.
// Scale (1/sqrt d) is applied inside the softmax pass.
// ---------------------------------------------------------------------------
__global__ __launch_bounds__(256) void scores_kernel(const unsigned short* __restrict__ qh,
                                                     const unsigned short* __restrict__ kh,
                                                     float* __restrict__ scores) {
    __shared__ __align__(16) unsigned short lA[128 * LPITCH];   // A[i][s]  (qh^T)
    __shared__ __align__(16) unsigned short lB[128 * LPITCH];   // Bt[j][s] (kh^T)

    const int bh = blockIdx.x;                       // b*8 + h
    const int b = bh >> 3, h = bh & 7;
    const int sBeg = blockIdx.y * (SEQ / 8);         // K-slice of 512
    const size_t base = (size_t)b * SEQ * HID + h * DCH;

    const int tid  = threadIdx.x;
    const int wave = tid >> 5;
    const int wm   = (wave & 3) * 32;                // i offset
    const int wn   = (wave >> 2) * 64;               // j offset

    v8f acc[2][4] = {};

    for (int s0 = sBeg; s0 < sBeg + SEQ / 8; s0 += 32) {
        #pragma unroll
        for (int e = 0; e < 8; ++e) {                // 2048 uint chunks (2 channels each)
            int idx = e * 256 + tid;
            int s = idx >> 6, i2 = (idx & 63) * 2;
            size_t g = base + (size_t)(s0 + s) * HID + i2;
            unsigned q2 = *(const unsigned*)(qh + g);
            unsigned k2 = *(const unsigned*)(kh + g);
            lA[i2 * LPITCH + s]       = (unsigned short)(q2 & 0xFFFF);
            lA[(i2 + 1) * LPITCH + s] = (unsigned short)(q2 >> 16);
            lB[i2 * LPITCH + s]       = (unsigned short)(k2 & 0xFFFF);
            lB[(i2 + 1) * LPITCH + s] = (unsigned short)(k2 >> 16);
        }
        __syncthreads();

        v16bf aF[2], bF[4];
        #pragma unroll
        for (int i = 0; i < 2; ++i) aF[i] = load_fragA(lA + (wm + i * 16) * LPITCH);
        #pragma unroll
        for (int j = 0; j < 4; ++j) bF[j] = load_fragB(lB + (wn + j * 16) * LPITCH);
        #pragma unroll
        for (int i = 0; i < 2; ++i)
            #pragma unroll
            for (int j = 0; j < 4; ++j)
                acc[i][j] = WMMA_BF16(aF[i], bF[j], acc[i][j]);
        __syncthreads();
    }

    const int lane = tid & 31;
    const int col = lane & 15;
    const int rowAdd = (lane >> 4) << 3;
    #pragma unroll
    for (int i = 0; i < 2; ++i)
        #pragma unroll
        for (int j = 0; j < 4; ++j)
            #pragma unroll
            for (int r = 0; r < 8; ++r) {
                int ii = wm + i * 16 + rowAdd + r;
                int jj = wn + j * 16 + col;
                atomicAdd(&scores[((size_t)bh * DCH + ii) * DCH + jj], acc[i][j][r]);
            }
}

// ---------------------------------------------------------------------------
// Kernel: row softmax over [B*H*D, 128] in place; applies 1/sqrt(d) scale.
// ---------------------------------------------------------------------------
__global__ __launch_bounds__(256) void softmax_rows(float* __restrict__ p, int rows) {
    const float scale = 0.08838834764831845f;        // 1/sqrt(128)
    int r = blockIdx.x * 256 + threadIdx.x;
    if (r >= rows) return;
    float* row = p + (size_t)r * DCH;
    float mx = -3.402823466e38f;
    for (int j = 0; j < DCH; ++j) mx = fmaxf(mx, row[j] * scale);
    float sum = 0.f;
    for (int j = 0; j < DCH; ++j) sum += __expf(row[j] * scale - mx);
    float inv = 1.0f / sum;
    for (int j = 0; j < DCH; ++j) row[j] = __expf(row[j] * scale - mx) * inv;
}

// ---------------------------------------------------------------------------
// Kernel: X2[(b,s), i*H+h] = sum_j attn[b,h,i,j] * vh[b,s,h,j]
// Grid: (bh, s-tile). Block tile 128(s) x 128(i), K = 128(j), 4 K-steps.
// vh staged async; attn (f32) staged with packed cvt.
// ---------------------------------------------------------------------------
__global__ __launch_bounds__(256) void ctx_kernel(const unsigned short* __restrict__ vh,
                                                  const float* __restrict__ attn,
                                                  unsigned short* __restrict__ X2) {
    __shared__ __align__(16) unsigned short lA[128 * LPITCH];   // A[s][j] (vh natural)
    __shared__ __align__(16) unsigned short lB[128 * LPITCH];   // Bt[i][j] (attn natural)

    const int bh = blockIdx.x;
    const int b = bh >> 3, h = bh & 7;
    const int s0 = blockIdx.y * 128;
    const size_t vbase = (size_t)b * SEQ * HID + h * DCH;
    const size_t abase = (size_t)bh * DCH * DCH;
    const unsigned lAbase = (unsigned)(size_t)(void*)lA;

    const int tid  = threadIdx.x;
    const int wave = tid >> 5;
    const int wm   = (wave & 3) * 32;                // s offset
    const int wn   = (wave >> 2) * 64;               // i offset

    v8f acc[2][4] = {};

    for (int j0 = 0; j0 < DCH; j0 += 32) {
        stage_async(lAbase, vh + vbase + (size_t)s0 * HID + j0, (size_t)HID, tid);
        #pragma unroll
        for (int e = 0; e < 4; ++e) {                // attn tile 128x32 f32 -> bf16
            int idx = e * 256 + tid;
            int r = idx >> 3, c = (idx & 7) * 4;
            float4 f = *(const float4*)(attn + abase + (size_t)r * DCH + j0 + c);
            uint2 p; p.x = pkbf(f.x, f.y); p.y = pkbf(f.z, f.w);
            *(uint2*)(lB + r * LPITCH + c) = p;
        }
        wait_async_all();
        __syncthreads();

        v16bf aF[2], bF[4];
        #pragma unroll
        for (int i = 0; i < 2; ++i) aF[i] = load_fragA(lA + (wm + i * 16) * LPITCH);
        #pragma unroll
        for (int j = 0; j < 4; ++j) bF[j] = load_fragB(lB + (wn + j * 16) * LPITCH);
        #pragma unroll
        for (int i = 0; i < 2; ++i)
            #pragma unroll
            for (int j = 0; j < 4; ++j)
                acc[i][j] = WMMA_BF16(aF[i], bF[j], acc[i][j]);
        __syncthreads();
    }

    const int lane = tid & 31;
    const int col = lane & 15;
    const int rowAdd = (lane >> 4) << 3;
    #pragma unroll
    for (int i = 0; i < 2; ++i)
        #pragma unroll
        for (int j = 0; j < 4; ++j)
            #pragma unroll
            for (int r = 0; r < 8; ++r) {
                int s  = wm + i * 16 + rowAdd + r;
                int ii = wn + j * 16 + col;
                X2[((size_t)b * SEQ + s0 + s) * HID + ii * NHEAD + h] = f2bf(acc[i][j][r]);
            }
}

// ---------------------------------------------------------------------------
// Launch: cvt(x7) -> 3 projections -> zero -> scores(split-K) -> softmax
//         -> context -> output projection
// ---------------------------------------------------------------------------
extern "C" void kernel_launch(void* const* d_in, const int* in_sizes, int n_in,
                              void* d_out, int out_size, void* d_ws, size_t ws_size,
                              hipStream_t stream) {
    const float* q  = (const float*)d_in[0];
    const float* k  = (const float*)d_in[1];
    const float* v  = (const float*)d_in[2];
    const float* Wq = (const float*)d_in[3];
    const float* Wk = (const float*)d_in[4];
    const float* Wv = (const float*)d_in[5];
    const float* Wo = (const float*)d_in[6];

    float* out  = (float*)d_out;                          // [B,S,HID] f32
    float* attn = out + (size_t)MROWS * HID;              // [B,H,D,D] f32

    const size_t BIG = (size_t)MROWS * HID;               // 16,777,216 elems
    const size_t WSZ = (size_t)HID * HID;                 // 1,048,576 elems
    unsigned short* ws  = (unsigned short*)d_ws;
    unsigned short* qb  = ws;                             // bf16 copies of inputs
    unsigned short* kb  = qb + BIG;
    unsigned short* vb  = kb + BIG;
    unsigned short* qh  = vb + BIG;                       // projected heads (bf16)
    unsigned short* kh  = qh + BIG;
    unsigned short* vh  = kh + BIG;
    unsigned short* wqb = vh + BIG;                       // bf16 weights
    unsigned short* wkb = wqb + WSZ;
    unsigned short* wvb = wkb + WSZ;
    unsigned short* wob = wvb + WSZ;
    unsigned short* X2  = qb;   // reuse: qb last read by the q-projection

    // 0) one-shot f32 -> bf16 conversions (streaming)
    const long n4i = (long)(BIG / 4), n4w = (long)(WSZ / 4);
    cvt_bf16<<<(unsigned)(n4i / 256), 256, 0, stream>>>(q,  qb,  n4i);
    cvt_bf16<<<(unsigned)(n4i / 256), 256, 0, stream>>>(k,  kb,  n4i);
    cvt_bf16<<<(unsigned)(n4i / 256), 256, 0, stream>>>(v,  vb,  n4i);
    cvt_bf16<<<(unsigned)(n4w / 256), 256, 0, stream>>>(Wq, wqb, n4w);
    cvt_bf16<<<(unsigned)(n4w / 256), 256, 0, stream>>>(Wk, wkb, n4w);
    cvt_bf16<<<(unsigned)(n4w / 256), 256, 0, stream>>>(Wv, wvb, n4w);
    cvt_bf16<<<(unsigned)(n4w / 256), 256, 0, stream>>>(Wo, wob, n4w);

    dim3 gGemm(HID / 128, MROWS / 128);                   // (8, 128)

    // 1) projections: qh/kh/vh = bf16(x @ W^T)   (async double-buffered)
    gemm_bf16<true><<<gGemm, 256, 0, stream>>>(qb, wqb, qh, MROWS, HID, HID);
    gemm_bf16<true><<<gGemm, 256, 0, stream>>>(kb, wkb, kh, MROWS, HID, HID);
    gemm_bf16<true><<<gGemm, 256, 0, stream>>>(vb, wvb, vh, MROWS, HID, HID);

    // 2) scores: zero accumulators, then split-K partial GEMMs + f32 atomics
    long nAttn4 = (long)(BATCH * NHEAD * DCH * DCH) / 4;
    zero_f32<<<(unsigned)((nAttn4 + 255) / 256), 256, 0, stream>>>(attn, nAttn4);
    scores_kernel<<<dim3(BATCH * NHEAD, 8), 256, 0, stream>>>(qh, kh, attn);

    // 3) softmax (applies 1/sqrt(d) scale)
    softmax_rows<<<(BATCH * NHEAD * DCH + 255) / 256, 256, 0, stream>>>(attn, BATCH * NHEAD * DCH);

    // 4) context, written permuted to [B,S,(d major, H minor)] as bf16
    ctx_kernel<<<dim3(BATCH * NHEAD, SEQ / 128), 256, 0, stream>>>(vh, attn, X2);

    // 5) output projection (f32 out)
    gemm_bf16<false><<<gGemm, 256, 0, stream>>>(X2, wob, out, MROWS, HID, HID);
}
